// VectorQuantizer_64785286693178
// MI455X (gfx1250) — matrix operand order repaired and checked by reading, not compiled, not run on hardware
//
#include <hip/hip_runtime.h>
#include <hip/hip_bf16.h>
#include <math.h>

// Problem dims (fixed by the reference)
#define BN   16
#define DD   256
#define HW   1024            // H*W
#define NN   16384           // B*H*W
#define KK   8192            // codebook size
#define ZQ_ELEMS  4194304    // B*D*H*W
#define OUT_LOSS  4194304
#define OUT_IDX   4194305
#define OUT_PERP  4210689

typedef float  v8f   __attribute__((ext_vector_type(8)));
typedef __bf16 v16bf __attribute__((ext_vector_type(16)));
typedef __bf16 v8bf  __attribute__((ext_vector_type(8)));
typedef int    v4i   __attribute__((ext_vector_type(4)));

#define AS1 __attribute__((address_space(1)))
#define AS3 __attribute__((address_space(3)))

// ---------------- workspace layout (bytes) ----------------
#define WS_ZBF    0u          // NN*DD bf16      = 8388608
#define WS_CBF    8388608u    // KK*DD bf16      = 4194304
#define WS_ZNORM  12582912u   // NN float        = 65536
#define WS_ENORM  12648448u   // KK float        = 32768
#define WS_IDX    12681216u   // NN int          = 65536
#define WS_CNT    12746752u   // KK int          = 32768
#define WS_PART   12779520u   // 64 float        = 256

// ---- CDNA5 async global->LDS copy (ASYNCcnt path), with safe fallbacks ----
#if defined(__has_builtin)
#if __has_builtin(__builtin_amdgcn_global_load_async_to_lds_b128)
#define HAVE_ASYNC_LDS 1
#endif
#if __has_builtin(__builtin_amdgcn_s_wait_asynccnt)
#define HAVE_WAIT_ASYNC_BUILTIN 1
#endif
#endif

__device__ __forceinline__ void async_copy_b128(const void* g, void* l) {
#ifdef HAVE_ASYNC_LDS
  __builtin_amdgcn_global_load_async_to_lds_b128(
      (AS1 v4i*)g, (AS3 v4i*)l, 0, 0);
#else
  *(uint4*)l = *(const uint4*)g;   // fallback: load + ds_store
#endif
}

__device__ __forceinline__ void wait_async_all() {
#ifdef HAVE_ASYNC_LDS
#ifdef HAVE_WAIT_ASYNC_BUILTIN
  __builtin_amdgcn_s_wait_asynccnt(0);
#else
  asm volatile("s_wait_asynccnt 0x0" ::: "memory");
#endif
#endif
}

__device__ __forceinline__ float blockReduceSum(float v, float* sh) {
  const int lane = threadIdx.x & 31;
  const int w    = threadIdx.x >> 5;
  const int nw   = (blockDim.x + 31) >> 5;
  #pragma unroll
  for (int off = 16; off; off >>= 1) v += __shfl_down(v, off, 32);
  __syncthreads();                 // sh may be reused across calls
  if (lane == 0) sh[w] = v;
  __syncthreads();
  v = 0.f;
  if (w == 0) {
    if (lane < nw) v = sh[lane];
    #pragma unroll
    for (int off = 4; off; off >>= 1) v += __shfl_down(v, off, 32);
  }
  return v;                        // valid on thread 0
}

// -------- 1. codebook f32 -> bf16, ||e||^2 (one row per block) --------
__global__ __launch_bounds__(256)
void prep_cb_kernel(const float* __restrict__ cb, __bf16* __restrict__ cbf,
                    float* __restrict__ enorm) {
  __shared__ float sh[8];
  const int r = blockIdx.x, d = threadIdx.x;
  float v = cb[(size_t)r * DD + d];
  cbf[(size_t)r * DD + d] = (__bf16)v;
  float s = blockReduceSum(v * v, sh);
  if (d == 0) enorm[r] = s;
}

// -------- 2. z_e (B,D,H,W) -> z_flat (N,D) bf16, ||z||^2 --------
__global__ __launch_bounds__(256)
void prep_z_kernel(const float* __restrict__ ze, __bf16* __restrict__ zbf,
                   float* __restrict__ znorm) {
  __shared__ float sh[8];
  const int n = blockIdx.x, d = threadIdx.x;
  const int b = n >> 10, hw = n & (HW - 1);
  float v = ze[((size_t)(b * DD + d) << 10) + hw];
  zbf[(size_t)n * DD + d] = (__bf16)v;
  float s = blockReduceSum(v * v, sh);
  if (d == 0) znorm[n] = s;
}

// -------- 3. zero histogram + loss partials --------
__global__ void zero_misc_kernel(int* __restrict__ counts, float* __restrict__ part) {
  const int g = blockIdx.x * blockDim.x + threadIdx.x;
  if (g < KK) counts[g] = 0;
  if (g < 64) part[g] = 0.f;
}

// -------- 4. distance GEMM (bf16 WMMA) + fused argmin --------
// block = 128 threads (4 wave32) -> 128 rows; each wave: 32 rows x all 8192 cols
// (two 16-row A groups reuse every B fragment -> 2 WMMAs per LDS B read).
// Codebook streamed through LDS in 64-col x 256-k bf16 tiles, double-buffered
// via GLOBAL_LOAD_ASYNC_TO_LDS_B128 one full tile (64 WMMAs) ahead.
#define ROWS_PB  128
#define COLS_PT  64
#define LDSROW   264        // 256 + 8 bf16 pad: 528B row stride rotates LDS banks
#define SM_ZT    0                                   // 128*264*2  = 67584
#define SM_ZNL   67584                               // 128*4      = 512
#define SM_B0    68096                               // 64*264*2   = 33792
#define SM_B1    (68096 + 33792)                     // 64*264*2   = 33792
#define SM_TOTAL (68096 + 2 * 33792)                 // 135680 bytes

__device__ __forceinline__ void stage_b_tile(const __bf16* __restrict__ cbf,
                                             __bf16* __restrict__ lbuf,
                                             int col0, int tid) {
  // 64 cols x 256 k bf16 = 2048 x 16B chunks; 128 threads x 16 chunks
  #pragma unroll
  for (int i = 0; i < 16; ++i) {
    const int c   = tid + i * 128;
    const int row = c >> 5, off = (c & 31) * 8;
    async_copy_b128(cbf + (size_t)(col0 + row) * DD + off,
                    lbuf + row * LDSROW + off);
  }
}

__global__ __launch_bounds__(128)
void vq_argmin_kernel(const __bf16* __restrict__ zbf, const __bf16* __restrict__ cbf,
                      const float* __restrict__ znorm, const float* __restrict__ enorm,
                      int* __restrict__ idx_out) {
  extern __shared__ __attribute__((aligned(16))) char smem[];
  __bf16* zt  = (__bf16*)(smem + SM_ZT);
  float*  znl = (float*)(smem + SM_ZNL);
  __bf16* b0  = (__bf16*)(smem + SM_B0);
  __bf16* b1  = (__bf16*)(smem + SM_B1);

  const int tid = threadIdx.x;
  const int rowBlock = blockIdx.x * ROWS_PB;

  // stage the 128x256 bf16 A-tile into LDS (regular loads, once)
  const uint4* src = (const uint4*)(zbf + (size_t)rowBlock * DD);
  for (int c = tid; c < ROWS_PB * 32; c += 128) {
    const int row = c >> 5, off = (c & 31) * 8;
    uint4 v = src[c];
    *(uint4*)&zt[row * LDSROW + off] = v;
  }
  for (int r = tid; r < ROWS_PB; r += 128) znl[r] = znorm[rowBlock + r];

  // kick off async stage of the first codebook tile
  stage_b_tile(cbf, b0, 0, tid);

  const int lane  = tid & 31;
  const int wave  = tid >> 5;
  const int hsel  = lane >> 4;      // lane half: K sub-block & C row half
  const int nIdx  = lane & 15;      // column-within-tile; A-row M for this lane
  const int rTile = wave * 32;      // 32 rows per wave (2 groups of 16)

  float bestd[2][8];
  int   besti[2][8];
  #pragma unroll
  for (int g = 0; g < 2; ++g)
    #pragma unroll
    for (int r = 0; r < 8; ++r) { bestd[g][r] = 3.4e38f; besti[g][r] = 0; }

  const __bf16* aRow0 = &zt[(rTile + nIdx) * LDSROW];
  const __bf16* aRow1 = &zt[(rTile + 16 + nIdx) * LDSROW];
  const v8f vzero = {0.f, 0.f, 0.f, 0.f, 0.f, 0.f, 0.f, 0.f};
  float zn[2][8];

  const int ntiles = KK / COLS_PT;  // 128
  for (int tile = 0; tile < ntiles; ++tile) {
    const int col0 = tile * COLS_PT;
    wait_async_all();               // this wave's async tile in LDS
    __syncthreads();                // everyone's tile (and prior compute) done
    if (tile == 0) {                // znl valid after first barrier
      #pragma unroll
      for (int g = 0; g < 2; ++g)
        #pragma unroll
        for (int r = 0; r < 8; ++r)
          zn[g][r] = znl[rTile + g * 16 + hsel * 8 + r];
    }
    if (tile + 1 < ntiles)          // overlap: next tile streams during compute
      stage_b_tile(cbf, ((tile + 1) & 1) ? b1 : b0, col0 + COLS_PT, tid);

    const __bf16* bb = (tile & 1) ? b1 : b0;
    const __bf16* bcol[4];
    float en[4];
    #pragma unroll
    for (int t = 0; t < 4; ++t) {
      bcol[t] = bb + (t * 16 + nIdx) * LDSROW + hsel * 8;
      en[t]   = enorm[col0 + t * 16 + nIdx];   // issued early, consumed late
    }

    v8f acc[2][4];
    #pragma unroll
    for (int t = 0; t < 4; ++t) { acc[0][t] = vzero; acc[1][t] = vzero; }

    #pragma unroll
    for (int k0 = 0; k0 < DD; k0 += 32) {
      // A fragments, ISA 16-bit 16x32 layout: half-wave K offsets {0,16} vs {8,24}
      v8bf lo0 = *(const v8bf*)(aRow0 + k0 + hsel * 8);
      v8bf hi0 = *(const v8bf*)(aRow0 + k0 + hsel * 8 + 16);
      v8bf lo1 = *(const v8bf*)(aRow1 + k0 + hsel * 8);
      v8bf hi1 = *(const v8bf*)(aRow1 + k0 + hsel * 8 + 16);
      v16bf a0 = __builtin_shufflevector(lo0, hi0, 0, 1, 2, 3, 4, 5, 6, 7,
                                                   8, 9, 10, 11, 12, 13, 14, 15);
      v16bf a1 = __builtin_shufflevector(lo1, hi1, 0, 1, 2, 3, 4, 5, 6, 7,
                                                   8, 9, 10, 11, 12, 13, 14, 15);
      #pragma unroll
      for (int t = 0; t < 4; ++t) {
        v16bf b = __builtin_shufflevector(*(const v8bf*)(bcol[t] + k0), *(const v8bf*)(bcol[t] + k0 + 16),
                                          0, 1, 2, 3, 4, 5, 6, 7, 8, 9, 10, 11, 12, 13, 14, 15);   // K offsets {0,16} vs {8,24}, 2 WMMAs
        acc[0][t] = __builtin_amdgcn_wmma_f32_16x16x32_bf16(
            false, a0, false, b, (short)0, acc[0][t], false, false);
        acc[1][t] = __builtin_amdgcn_wmma_f32_16x16x32_bf16(
            false, a1, false, b, (short)0, acc[1][t], false, false);
      }
    }
    // dist = ||z||^2 + ||e||^2 - 2*dot ; fused running argmin
    #pragma unroll
    for (int t = 0; t < 4; ++t) {
      const int col = col0 + t * 16 + nIdx;
      #pragma unroll
      for (int g = 0; g < 2; ++g)
        #pragma unroll
        for (int r = 0; r < 8; ++r) {
          float d = zn[g][r] + en[t] - 2.0f * acc[g][t][r];
          if (d < bestd[g][r]) { bestd[g][r] = d; besti[g][r] = col; }
        }
    }
    __syncthreads();                // compute done before next overwrite of bb
  }

  // reduce across the 16 lanes that share the same rows (first-min index wins)
  #pragma unroll
  for (int off = 8; off >= 1; off >>= 1) {
    #pragma unroll
    for (int g = 0; g < 2; ++g)
      #pragma unroll
      for (int r = 0; r < 8; ++r) {
        float od = __shfl_xor(bestd[g][r], off, 16);
        int   oi = __shfl_xor(besti[g][r], off, 16);
        if (od < bestd[g][r] || (od == bestd[g][r] && oi < besti[g][r])) {
          bestd[g][r] = od; besti[g][r] = oi;
        }
      }
  }
  if (nIdx == 0) {
    #pragma unroll
    for (int g = 0; g < 2; ++g)
      #pragma unroll
      for (int r = 0; r < 8; ++r)
        idx_out[rowBlock + rTile + g * 16 + hsel * 8 + r] = besti[g][r];
  }
}

// -------- 5. gather z_q, write (B,D,H,W), SSE partials, histogram, indices --------
__global__ __launch_bounds__(256)
void gather_out_kernel(const float* __restrict__ cb, const float* __restrict__ ze,
                       const int* __restrict__ idx_ws, float* __restrict__ out,
                       int* __restrict__ counts, float* __restrict__ part) {
  __shared__ float sh[8];
  const int n = blockIdx.x * 256 + threadIdx.x;
  const int b = n >> 10, hw = n & (HW - 1);
  const int idx = idx_ws[n];
  const float* crow = cb + (size_t)idx * DD;
  float sse = 0.f;
  #pragma unroll 4
  for (int d = 0; d < DD; ++d) {
    const size_t off = ((size_t)(b * DD + d) << 10) + hw;   // coalesced across block
    float zq = crow[d];
    float zv = ze[off];
    out[off] = zq;                  // z_q_st forward value == z_q
    float df = zq - zv;
    sse += df * df;
  }
  atomicAdd(&counts[idx], 1);
  out[OUT_IDX + n] = (float)idx;
  sse = blockReduceSum(sse, sh);
  if (threadIdx.x == 0) part[blockIdx.x] = sse;
}

// -------- 6. finalize: vq_loss + perplexity --------
__global__ __launch_bounds__(256)
void finalize_kernel(const int* __restrict__ counts, const float* __restrict__ part,
                     float* __restrict__ out) {
  __shared__ float sh[8];
  __shared__ float vq_s;
  const int tid = threadIdx.x;
  float s = (tid < 64) ? part[tid] : 0.f;
  s = blockReduceSum(s, sh);
  if (tid == 0) vq_s = 1.25f * (s / (float)ZQ_ELEMS);   // (1 + BETA) * MSE
  float h = 0.f;
  for (int k = tid; k < KK; k += 256) {
    float p = (float)counts[k] * (1.0f / (float)NN);
    h += p * logf(p + 1e-10f);
  }
  h = blockReduceSum(h, sh);
  if (tid == 0) {
    out[OUT_LOSS] = vq_s;
    out[OUT_PERP] = expf(-h);
  }
}

extern "C" void kernel_launch(void* const* d_in, const int* in_sizes, int n_in,
                              void* d_out, int out_size, void* d_ws, size_t ws_size,
                              hipStream_t stream) {
  const float* z_e = (const float*)d_in[0];     // (16,256,32,32) f32
  const float* cb  = (const float*)d_in[1];     // (8192,256) f32
  float* out = (float*)d_out;

  char* ws = (char*)d_ws;
  __bf16* zbf   = (__bf16*)(ws + WS_ZBF);
  __bf16* cbf   = (__bf16*)(ws + WS_CBF);
  float*  znorm = (float*)(ws + WS_ZNORM);
  float*  enorm = (float*)(ws + WS_ENORM);
  int*    idx   = (int*)(ws + WS_IDX);
  int*    cnt   = (int*)(ws + WS_CNT);
  float*  part  = (float*)(ws + WS_PART);

  prep_cb_kernel<<<KK, 256, 0, stream>>>(cb, cbf, enorm);
  prep_z_kernel<<<NN, 256, 0, stream>>>(z_e, zbf, znorm);
  zero_misc_kernel<<<(KK + 255) / 256, 256, 0, stream>>>(cnt, part);
  vq_argmin_kernel<<<NN / ROWS_PB, 128, SM_TOTAL, stream>>>(zbf, cbf, znorm, enorm, idx);
  gather_out_kernel<<<NN / 256, 256, 0, stream>>>(cb, z_e, idx, out, cnt, part);
  finalize_kernel<<<1, 256, 0, stream>>>(cnt, part, out);
}
